// CompressedMultiHeadAttention_86114094285131
// MI455X (gfx1250) — matrix operand order repaired and checked by reading, not compiled, not run on hardware
//
#include <hip/hip_runtime.h>

// ---------------------------------------------------------------------------
// GQA attention layer for gfx1250 (MI455X), built on v_wmma_f32_16x16x32_f16
// plus async global->LDS DMA (global_load_async_to_lds_b128 / ASYNCcnt).
//
// Shapes: B=1, S=4096, E=2048, H=16, HK=4, D=128 (rep=4).
// Pipeline:
//   1) convert x, Wq, Wk, Wv, Wo to fp16 in workspace
//   2) Q  = x @ Wq^T   (fp16 out, [S, H*D])          -- WMMA GEMM
//   3) K  = x @ Wk^T   (fp16 out, [S, HK*D])         -- WMMA GEMM
//   4) Vt = (x @ Wv^T)^T (fp16 out, [HK*D, S])       -- WMMA GEMM, transposed store
//   5) flash attention: K/V tiles double-buffered into LDS via async-to-LDS
//      DMA, scores computed as S^T = K Q^T (WMMA), PV as out^T = V^T P^T
//   6) out = attn @ Wo^T (fp32 out)                   -- WMMA GEMM
// ---------------------------------------------------------------------------

typedef _Float16 v16h __attribute__((ext_vector_type(16)));
typedef _Float16 v8h  __attribute__((ext_vector_type(8)));
typedef float    v8f  __attribute__((ext_vector_type(8)));

#define S_LEN 4096
#define E_DIM 2048
#define NH    16
#define NKV   4
#define HD    128

#define KROW  136   // K tile LDS row stride in halves (272B: bank start 4r, conflict-free)
#define VROW  40    // V / P^T LDS row stride in halves (80B: bank start 20r, conflict-free)

static __device__ __forceinline__ v8f wmma_f16(v16h a, v16h b, v8f c) {
  // D = A(16x32 f16) * B(32x16 f16) + C(16x16 f32)
  return __builtin_amdgcn_wmma_f32_16x16x32_f16(
      /*neg_a=*/false, a, /*neg_b=*/false, b,
      /*c_mod=*/(short)0, c, /*reuse_a=*/false, /*reuse_b=*/false);
}

// Load two contiguous 16-byte chunks into one 16-element f16 fragment.
static __device__ __forceinline__ v16h ldpair(const _Float16* p0, const _Float16* p1) {
  v8h a = *reinterpret_cast<const v8h*>(p0);
  v8h b = *reinterpret_cast<const v8h*>(p1);
  return __builtin_shufflevector(a, b, 0, 1, 2, 3, 4, 5, 6, 7,
                                       8, 9, 10, 11, 12, 13, 14, 15);
}

// Async DMA of one 16-byte chunk global -> LDS (tracked by ASYNCcnt).
static __device__ __forceinline__ void async_ld16(unsigned lds, const _Float16* g) {
  asm volatile("global_load_async_to_lds_b128 %0, %1, off"
               :: "v"(lds), "v"(g) : "memory");
}

// ---------------------------------------------------------------------------
// fp32 -> fp16 conversion, 4 elements/thread
// ---------------------------------------------------------------------------
__global__ void cvt_f32_f16(const float* __restrict__ s, _Float16* __restrict__ d,
                            int nElems) {
  int i = (blockIdx.x * blockDim.x + threadIdx.x) * 4;
  if (i + 3 < nElems) {
    float4 v = *reinterpret_cast<const float4*>(s + i);
    _Float16* o = d + i;
    o[0] = (_Float16)v.x; o[1] = (_Float16)v.y;
    o[2] = (_Float16)v.z; o[3] = (_Float16)v.w;
  }
}

// ---------------------------------------------------------------------------
// C[m,n] = sum_k A[m,k] * Bw[n,k]   (both operands row-major over k — the
// natural Linear layout). Each wave computes a 16x64 tile (A-frag reused 4x).
// Block = 256 threads = 8 waves -> 128 rows x 64 cols per block. Weights fit
// the 192MB L2, so per-wave streaming re-hits L2, not HBM.
// ---------------------------------------------------------------------------
template <typename OutT>
__global__ __launch_bounds__(256) void gemm_nt_wmma(
    const _Float16* __restrict__ A, const _Float16* __restrict__ Bw,
    OutT* __restrict__ C, int Kd, long ldcM, long ldcN) {
  const int tid  = threadIdx.x;
  const int lane = tid & 31;
  const int w    = tid >> 5;
  const int g    = lane >> 4;  // lane half: selects K sub-range of fragments
  const int r    = lane & 15;  // A row within tile / B column within tile
  const int mBase = blockIdx.y * 128 + w * 16;
  const int nBase = blockIdx.x * 64;

  v8f acc[4];
#pragma unroll
  for (int t = 0; t < 4; ++t)
#pragma unroll
    for (int i = 0; i < 8; ++i) acc[t][i] = 0.0f;

  const _Float16* arow = A + (size_t)(mBase + r) * Kd;
  for (int kk = 0; kk < Kd; kk += 32) {
    __builtin_prefetch(arow + kk + 256, 0, 0);  // global_prefetch_b8
    // A-frag (16x32 f16): lane half g holds K = g*8..g*8+7 and 16+g*8..+7
    v16h a = ldpair(arow + kk + g * 8, arow + kk + 16 + g * 8);
#pragma unroll
    for (int t = 0; t < 4; ++t) {
      // B-frag (32x16): lane half g holds K = g*8..g*8+7 and 16+g*8..+7, column n=r
      const _Float16* bp = Bw + (size_t)(nBase + t * 16 + r) * Kd + kk + g * 8;
      v16h b = ldpair(bp, bp + 16);
      acc[t] = wmma_f16(a, b, acc[t]);
    }
  }
#pragma unroll
  for (int t = 0; t < 4; ++t) {
    const long n = nBase + t * 16 + r;
#pragma unroll
    for (int i = 0; i < 8; ++i) {
      const long m = mBase + 8 * g + i;  // C layout: vgpr i, half g -> row i+8g
      C[m * ldcM + n * ldcN] = (OutT)acc[t][i];
    }
  }
}

// ---------------------------------------------------------------------------
// Flash attention. Block = 128 threads = 4 waves, all on the SAME head with
// four different 16-query tiles, so K/V tiles are shared: they are staged
// into LDS once per block with global_load_async_to_lds_b128 (double
// buffered; next tile's DMA overlaps current tile's WMMA), instead of each
// wave re-streaming them 4x from L2.
//   S^T = K * Q^T  : A = K rows (LDS), B = Q rows (registers, preloaded).
//   C-layout of S^T puts query q in lane column q -> softmax reductions are
//   in-lane over 8 VGPRs + one shfl_xor(16).
//   out^T = V^T * P^T : A = Vt rows (LDS), B = P^T via 1KB/wave LDS
//   round-trip (2x b128 store/load).
// ---------------------------------------------------------------------------
__global__ __launch_bounds__(128) void flash_attn_wmma(
    const _Float16* __restrict__ Q, const _Float16* __restrict__ Kb,
    const _Float16* __restrict__ Vt, _Float16* __restrict__ O) {
  __shared__ __align__(16) _Float16 kbuf[2][32 * KROW];   // 2 x 8704B
  __shared__ __align__(16) _Float16 vbuf[2][128 * VROW];  // 2 x 10240B
  __shared__ __align__(16) _Float16 ptb[4][16 * VROW];    // per-wave P^T scratch

  const int tid  = threadIdx.x;
  const int lane = tid & 31;
  const int w    = tid >> 5;
  const int g    = lane >> 4;
  const int n    = lane & 15;   // query column (B/C frags) and row index (A frags)
  const int hq   = blockIdx.y;  // query head
  const int kh   = hq >> 2;     // kv head (rep = 4)
  const int qBase = (blockIdx.x * 4 + w) * 16;
  _Float16* pt = ptb[w];

  const _Float16* Kg = Kb + (size_t)kh * HD;              // + row*512 later
  const _Float16* Vg = Vt + (size_t)(kh * HD) * S_LEN;    // + row*4096 later

  // Stage one 32-key block: K tile 32x128 f16 (8KB) + V^T tile 128x32 f16
  // (8KB); 512+512 16B chunks, 8 async DMA instructions per thread.
  auto stage = [&](int buf, int kb2) {
#pragma unroll
    for (int j = 0; j < 4; ++j) {
      const int c  = tid + j * 128;          // 0..511
      const int kr = c >> 3, ko = (c & 7) * 8;
      async_ld16((unsigned)(size_t)(&kbuf[buf][kr * KROW + ko]),
                 Kg + (size_t)(kb2 + kr) * (NKV * HD) + ko);
      const int vr = c >> 2, vo = (c & 3) * 8;
      async_ld16((unsigned)(size_t)(&vbuf[buf][vr * VROW + vo]),
                 Vg + (size_t)vr * S_LEN + kb2 + vo);
    }
  };

  // Preload Q^T B-fragments for all four 32-wide d-steps (reused every key block)
  v16h qb[4];
#pragma unroll
  for (int ds = 0; ds < 4; ++ds) {
    const _Float16* qp = Q + (size_t)(qBase + n) * (NH * HD) + hq * HD + ds * 32 + g * 8;
    qb[ds] = ldpair(qp, qp + 16);
  }

  v8f ot[8];  // out^T accumulators: 8 d-tiles x (16 d x 16 q)
#pragma unroll
  for (int f = 0; f < 8; ++f)
#pragma unroll
    for (int i = 0; i < 8; ++i) ot[f][i] = 0.0f;

  float m_run = -3.0e38f, l_run = 0.0f;
  const float scale = 0.08838834764831845f;  // 1/sqrt(128)

  stage(0, 0);  // prologue DMA for tile 0

  const int nTiles = S_LEN / 32;
  for (int it = 0; it < nTiles; ++it) {
    const int cur = it & 1;
    // All waves done READING buf[1-cur] (their iter it-1 tile) before new DMA.
    __syncthreads();
    if (it + 1 < nTiles) {
      stage(1 - cur, (it + 1) * 32);
      // 8 DMAs outstanding for next tile; retire current tile's 8 (in-order).
      asm volatile("s_wait_asynccnt 8" ::: "memory");
    } else {
      asm volatile("s_wait_asynccnt 0" ::: "memory");
    }
    __syncthreads();  // tile `it` fully resident across all waves
    const _Float16* kc = kbuf[cur];
    const _Float16* vc = vbuf[cur];

    // ---- scores: S^T two 16-key fragments, accumulate over d in 4 WMMA steps
    v8f s0, s1;
#pragma unroll
    for (int i = 0; i < 8; ++i) { s0[i] = 0.0f; s1[i] = 0.0f; }
#pragma unroll
    for (int ds = 0; ds < 4; ++ds) {
      const _Float16* k0 = kc + n * KROW + ds * 32;
      s0 = wmma_f16(ldpair(k0 + g * 8, k0 + 16 + g * 8), qb[ds], s0);
      const _Float16* k1 = kc + (16 + n) * KROW + ds * 32;
      s1 = wmma_f16(ldpair(k1 + g * 8, k1 + 16 + g * 8), qb[ds], s1);
    }
    // ---- online softmax (per query = per lane column)
    float vm = -3.0e38f;
#pragma unroll
    for (int i = 0; i < 8; ++i) {
      s0[i] = s0[i] * scale;
      s1[i] = s1[i] * scale;
      vm = fmaxf(vm, fmaxf(s0[i], s1[i]));
    }
    vm = fmaxf(vm, __shfl_xor(vm, 16, 32));  // other 16 keys live in lane^16
    const float m_new = fmaxf(m_run, vm);
    const float alpha = __expf(m_run - m_new);
    float rsum = 0.0f;
    v8h p0, p1;
#pragma unroll
    for (int i = 0; i < 8; ++i) {
      float e0 = __expf(s0[i] - m_new);
      float e1 = __expf(s1[i] - m_new);
      rsum += e0 + e1;
      p0[i] = (_Float16)e0;
      p1[i] = (_Float16)e1;
    }
    rsum += __shfl_xor(rsum, 16, 32);
    l_run = l_run * alpha + rsum;
    m_run = m_new;
#pragma unroll
    for (int f = 0; f < 8; ++f)
#pragma unroll
      for (int i = 0; i < 8; ++i) ot[f][i] *= alpha;

    // ---- relayout P (C-layout) -> P^T B-fragment via LDS: row q, k contiguous
    *reinterpret_cast<v8h*>(pt + n * VROW + 8 * g)      = p0;  // keys 8g..8g+7
    *reinterpret_cast<v8h*>(pt + n * VROW + 16 + 8 * g) = p1;  // keys 16+8g..+7
    asm volatile("s_wait_dscnt 0" ::: "memory");
    v16h pb = ldpair(pt + n * VROW + g * 8, pt + n * VROW + 16 + g * 8);

    // ---- out^T += V^T(16d x 32k) * P^T(32k x 16q), 8 d-tiles from LDS
#pragma unroll
    for (int dt = 0; dt < 8; ++dt) {
      const _Float16* vp = vc + (dt * 16 + n) * VROW;
      ot[dt] = wmma_f16(ldpair(vp + g * 8, vp + 16 + g * 8), pb, ot[dt]);
    }
  }

  // ---- normalize and store: lane holds 8 contiguous d for query (qBase+n)
  const float inv = 1.0f / l_run;
#pragma unroll
  for (int dt = 0; dt < 8; ++dt) {
    v8h oh;
#pragma unroll
    for (int i = 0; i < 8; ++i) oh[i] = (_Float16)(ot[dt][i] * inv);
    *reinterpret_cast<v8h*>(O + (size_t)(qBase + n) * (NH * HD) + hq * HD +
                            dt * 16 + 8 * g) = oh;
  }
}

// ---------------------------------------------------------------------------
extern "C" void kernel_launch(void* const* d_in, const int* in_sizes, int n_in,
                              void* d_out, int out_size, void* d_ws, size_t ws_size,
                              hipStream_t stream) {
  (void)in_sizes; (void)n_in; (void)out_size; (void)ws_size;
  const float* x  = (const float*)d_in[0];
  const float* Wq = (const float*)d_in[1];
  const float* Wk = (const float*)d_in[2];
  const float* Wv = (const float*)d_in[3];
  const float* Wo = (const float*)d_in[4];

  char* ws = (char*)d_ws;
  size_t off = 0;
  auto alloc = [&](size_t bytes) -> void* {
    void* p = ws + off;
    off += (bytes + 255) & ~(size_t)255;
    return p;
  };
  _Float16* xh  = (_Float16*)alloc((size_t)S_LEN * E_DIM * 2);
  _Float16* wqh = (_Float16*)alloc((size_t)NH * HD * E_DIM * 2);
  _Float16* wkh = (_Float16*)alloc((size_t)NKV * HD * E_DIM * 2);
  _Float16* wvh = (_Float16*)alloc((size_t)NKV * HD * E_DIM * 2);
  _Float16* woh = (_Float16*)alloc((size_t)E_DIM * NH * HD * 2);
  _Float16* Qb  = (_Float16*)alloc((size_t)S_LEN * NH * HD * 2);
  _Float16* Kbf = (_Float16*)alloc((size_t)S_LEN * NKV * HD * 2);
  _Float16* Vtb = (_Float16*)alloc((size_t)NKV * HD * S_LEN * 2);  // [HK*D, S]
  _Float16* AO  = (_Float16*)alloc((size_t)S_LEN * NH * HD * 2);

  auto cvt = [&](const float* src, _Float16* dst, int nElems) {
    int threads = nElems / 4;
    cvt_f32_f16<<<(threads + 255) / 256, 256, 0, stream>>>(src, dst, nElems);
  };
  cvt(x,  xh,  S_LEN * E_DIM);
  cvt(Wq, wqh, NH * HD * E_DIM);
  cvt(Wk, wkh, NKV * HD * E_DIM);
  cvt(Wv, wvh, NKV * HD * E_DIM);
  cvt(Wo, woh, E_DIM * NH * HD);

  // Q = x @ Wq^T : [S, H*D] fp16
  gemm_nt_wmma<_Float16><<<dim3((NH * HD) / 64, S_LEN / 128), 256, 0, stream>>>(
      xh, wqh, Qb, E_DIM, (long)(NH * HD), 1L);
  // K = x @ Wk^T : [S, HK*D] fp16
  gemm_nt_wmma<_Float16><<<dim3((NKV * HD) / 64, S_LEN / 128), 256, 0, stream>>>(
      xh, wkh, Kbf, E_DIM, (long)(NKV * HD), 1L);
  // Vt = (x @ Wv^T)^T : [HK*D, S] fp16 (transposed store via strides)
  gemm_nt_wmma<_Float16><<<dim3((NKV * HD) / 64, S_LEN / 128), 256, 0, stream>>>(
      xh, wvh, Vtb, E_DIM, 1L, (long)S_LEN);

  // attention core (WMMA + async-to-LDS double buffering)
  flash_attn_wmma<<<dim3(S_LEN / 64, NH), 128, 0, stream>>>(Qb, Kbf, Vtb, AO);

  // out = attn @ Wo^T : fp32
  gemm_nt_wmma<float><<<dim3(E_DIM / 64, S_LEN / 128), 256, 0, stream>>>(
      AO, woh, (float*)d_out, NH * HD, (long)E_DIM, 1L);
}
